// PointNet2_52321291600081
// MI455X (gfx1250) — matrix operand order repaired and checked by reading, not compiled, not run on hardware
//
#include <hip/hip_runtime.h>

// ---------------------------------------------------------------------------
// Types for WMMA (gfx1250, wave32)
// ---------------------------------------------------------------------------
typedef __attribute__((ext_vector_type(16))) _Float16 v16h;
typedef __attribute__((ext_vector_type(8)))  _Float16 v8h;
typedef __attribute__((ext_vector_type(8)))  float    v8f;

// ---------------------------------------------------------------------------
// Staging loads for one 32(M)x32(K) A panel + 32(K)x64(N) B panel.
// Fast path: 16B vector loads. Slow path: hoisted row pointer, 32-bit
// clamped column index, value select (no per-element 64-bit address math).
// ---------------------------------------------------------------------------
__device__ __forceinline__ void stage_load(
    const _Float16* __restrict__ Ab, const _Float16* __restrict__ Bb,
    int lda, int ldb, int M, int N, int K,
    int m0, int n0, int k0,
    int ar, int ac0, int br, int bc0,
    bool alignA, bool alignB,
    v8h& ra, v8h& rb0, v8h& rb1)
{
    if (alignA && (m0 + 32 <= M) && (k0 + 32 <= K)) {
        ra = *(const v8h*)(Ab + (long long)(m0 + ar) * lda + (k0 + ac0));
    } else {
        int m = m0 + ar;
        bool mok = m < M;
        const _Float16* rowp = Ab + (long long)(mok ? m : 0) * lda;
#pragma unroll
        for (int j = 0; j < 8; ++j) {
            int k = k0 + ac0 + j;
            bool ok = mok && (k < K);
            _Float16 v = rowp[ok ? k : 0];
            ra[j] = ok ? v : (_Float16)0.0f;
        }
    }
    if (alignB && (n0 + 64 <= N) && (k0 + 32 <= K)) {
        long long rowoff = (long long)(k0 + br) * ldb + n0 + bc0;
        if (k0 + 64 <= K)                         // hint next B panel
            __builtin_prefetch(Bb + rowoff + 32LL * ldb, 0, 1);
        rb0 = *(const v8h*)(Bb + rowoff);
        rb1 = *(const v8h*)(Bb + rowoff + 8);
    } else {
        int k = k0 + br;
        bool kok = k < K;
        const _Float16* rowp = Bb + (long long)(kok ? k : 0) * ldb;
#pragma unroll
        for (int j = 0; j < 8; ++j) {
            int n = n0 + bc0 + j;
            bool ok = kok && (n < N);
            _Float16 v = rowp[ok ? n : 0];
            rb0[j] = ok ? v : (_Float16)0.0f;
        }
#pragma unroll
        for (int j = 0; j < 8; ++j) {
            int n = n0 + bc0 + 8 + j;
            bool ok = kok && (n < N);
            _Float16 v = rowp[ok ? n : 0];
            rb1[j] = ok ? v : (_Float16)0.0f;
        }
    }
}

// ---------------------------------------------------------------------------
// Generic fused GEMM:  Y[m][n] = act( scale[m] * (sum_k A[m][k]*B[k][n]) + bias[m] )
// A: f16 row-major [M][K] (lda), B: f16 row-major [K][N] (ldb).
// Batched via blockIdx.z with element strides bsA/bsB/bsC.
// Block = 128 threads (4 waves); tile = 32(M) x 64(N); K in chunks of 32,
// double-buffered through LDS (one barrier per chunk, global loads of
// chunk i+1 overlap the WMMAs of chunk i).
// Each wave computes two 16x16 WMMA tiles (M split), sharing one B fragment.
// LDS: A row-major (16B-aligned 8-half runs match the ISA A fragment layout),
//      B stored transposed so B fragments are two contiguous ds_load_b128.
// act: 0=none, 1=relu, 2=leaky(0.2)
// ---------------------------------------------------------------------------
__global__ __launch_bounds__(128)
void k_gemm_wmma(const _Float16* __restrict__ A, int lda, long long bsA,
                 const _Float16* __restrict__ Bm, int ldb, long long bsB,
                 float* __restrict__ Cf, _Float16* __restrict__ Ch,
                 int ldc, long long bsC,
                 int M, int N, int K,
                 const float* __restrict__ scale, const float* __restrict__ bias,
                 int act)
{
    __shared__ alignas(16) _Float16 sA[2][32][40];   // 32 rows x 32 K (stride 80B)
    __shared__ alignas(16) _Float16 sBt[2][64][40];  // 64 cols x 32 K (transposed)

    const int tid  = threadIdx.x;
    const int wave = tid >> 5;
    const int lane = tid & 31;
    const int m0   = blockIdx.y * 32;
    const int n0   = blockIdx.x * 64;
    const int b    = blockIdx.z;

    const _Float16* Ab = A  + (long long)b * bsA;
    const _Float16* Bb = Bm + (long long)b * bsB;

    const int ar  = tid >> 2;         // 0..31 : A row within tile
    const int ac0 = (tid & 3) * 8;    // A col group (8 halves)
    const int br  = tid >> 2;         // 0..31 : B k-row within tile
    const int bc0 = (tid & 3) * 16;   // B col group (16 halves)

    const bool alignA = ((lda & 7) == 0) && ((bsA & 7) == 0);
    const bool alignB = ((ldb & 7) == 0) && ((bsB & 7) == 0);

    // fragment addressing (ISA 7.12.2):
    // A: lanes 0-15 row=lane, K {kb..kb+7, kb+16..kb+23} with kb=0;
    //    lanes 16-31 same rows, kb=8.
    const int arow = lane & 15;
    const int kb   = (lane >> 4) * 8;
    const int bcol = (lane & 15) + wave * 16;
    const int kbb  = (lane >> 4) * 8;

    v8f acc0 = {};
    v8f acc1 = {};
    v8h ra, rb0, rb1;

    // prologue: stage chunk 0 into buffer 0
    stage_load(Ab, Bb, lda, ldb, M, N, K, m0, n0, 0,
               ar, ac0, br, bc0, alignA, alignB, ra, rb0, rb1);
    *(v8h*)&sA[0][ar][ac0] = ra;
#pragma unroll
    for (int j = 0; j < 8; ++j) sBt[0][bc0 + j][br] = rb0[j];
#pragma unroll
    for (int j = 0; j < 8; ++j) sBt[0][bc0 + 8 + j][br] = rb1[j];
    __syncthreads();

    int cur = 0;
    for (int k0 = 0; k0 < K; k0 += 32) {
        const bool have_next = (k0 + 32) < K;
        if (have_next)
            stage_load(Ab, Bb, lda, ldb, M, N, K, m0, n0, k0 + 32,
                       ar, ac0, br, bc0, alignA, alignB, ra, rb0, rb1);

        // ---- fragments (all 128-bit LDS reads) --------------------------
        v8h a0lo = *(const v8h*)&sA[cur][arow][kb];
        v8h a0hi = *(const v8h*)&sA[cur][arow][kb + 16];
        v8h a1lo = *(const v8h*)&sA[cur][16 + arow][kb];
        v8h a1hi = *(const v8h*)&sA[cur][16 + arow][kb + 16];
        v8h blo  = *(const v8h*)&sBt[cur][bcol][kbb];
        v8h bhi  = *(const v8h*)&sBt[cur][bcol][kbb + 16];

        v16h av0 = __builtin_shufflevector(a0lo, a0hi,
                     0,1,2,3,4,5,6,7,8,9,10,11,12,13,14,15);
        v16h av1 = __builtin_shufflevector(a1lo, a1hi,
                     0,1,2,3,4,5,6,7,8,9,10,11,12,13,14,15);
        v16h bv  = __builtin_shufflevector(blo, bhi,
                     0,1,2,3,4,5,6,7,8,9,10,11,12,13,14,15);

        acc0 = __builtin_amdgcn_wmma_f32_16x16x32_f16(false, av0, false, bv,
                                                      (short)0, acc0, false, false);
        acc1 = __builtin_amdgcn_wmma_f32_16x16x32_f16(false, av1, false, bv,
                                                      (short)0, acc1, false, false);

        if (have_next) {
            int nxt = cur ^ 1;
            *(v8h*)&sA[nxt][ar][ac0] = ra;
#pragma unroll
            for (int j = 0; j < 8; ++j) sBt[nxt][bc0 + j][br] = rb0[j];
#pragma unroll
            for (int j = 0; j < 8; ++j) sBt[nxt][bc0 + 8 + j][br] = rb1[j];
        }
        __syncthreads();
        cur ^= 1;
    }

    // ---- store: C/D layout: vgpr r -> M = r + 8*(lane>=16), N = lane&15 ----
    const int nn = n0 + wave * 16 + (lane & 15);
    v8f accs[2] = {acc0, acc1};
    if ((m0 + 32 <= M) && (nn < N)) {
#pragma unroll
        for (int t2 = 0; t2 < 2; ++t2) {
            int mbase = m0 + 16 * t2 + ((lane >> 4) ? 8 : 0);
#pragma unroll
            for (int r = 0; r < 8; ++r) {
                int mm = mbase + r;
                float y = accs[t2][r];
                float s = scale ? scale[mm] : 1.0f;
                float bb = bias ? bias[mm] : 0.0f;
                y = y * s + bb;
                if (act == 1)      y = fmaxf(y, 0.0f);
                else if (act == 2) y = (y > 0.0f) ? y : 0.2f * y;
                long long ci = (long long)b * bsC + (long long)mm * ldc + nn;
                if (Cf) Cf[ci] = y;
                if (Ch) Ch[ci] = (_Float16)y;
            }
        }
    } else {
#pragma unroll
        for (int t2 = 0; t2 < 2; ++t2) {
            int mbase = m0 + 16 * t2 + ((lane >> 4) ? 8 : 0);
#pragma unroll
            for (int r = 0; r < 8; ++r) {
                int mm = mbase + r;
                if (mm < M && nn < N) {
                    float y = accs[t2][r];
                    float s = scale ? scale[mm] : 1.0f;
                    float bb = bias ? bias[mm] : 0.0f;
                    y = y * s + bb;
                    if (act == 1)      y = fmaxf(y, 0.0f);
                    else if (act == 2) y = (y > 0.0f) ? y : 0.2f * y;
                    long long ci = (long long)b * bsC + (long long)mm * ldc + nn;
                    if (Cf) Cf[ci] = y;
                    if (Ch) Ch[ci] = (_Float16)y;
                }
            }
        }
    }
}

// ---------------------------------------------------------------------------
// Utility kernels
// ---------------------------------------------------------------------------
__global__ void k_cvt16(const float* __restrict__ a, _Float16* __restrict__ o, long long n)
{
    long long i = (long long)blockIdx.x * blockDim.x + threadIdx.x;
    if (i < n) o[i] = (_Float16)a[i];
}

// fold BN + conv bias:  s = g/sqrt(1+eps) (or 1);  b' = s*cb + bn_b
__global__ void k_prep(const float* g, const float* bnb, const float* cb,
                       float* s, float* bias, int C)
{
    int i = blockIdx.x * blockDim.x + threadIdx.x;
    if (i >= C) return;
    float sc = g ? g[i] * rsqrtf(1.0f + 1e-5f) : 1.0f;
    float bb = (cb ? sc * cb[i] : 0.0f) + (bnb ? bnb[i] : 0.0f);
    s[i] = sc; bias[i] = bb;
}

// points [b][3][n] -> xyz [b][n][3]
__global__ void k_pts2xyz(const float* __restrict__ pts, float* __restrict__ xyz, int n)
{
    long long t = (long long)blockIdx.x * blockDim.x + threadIdx.x;
    if (t >= 8LL * n) return;
    int bq = (int)(t / n); long long i = t % n;
#pragma unroll
    for (int k = 0; k < 3; ++k)
        xyz[t * 3 + k] = pts[((long long)bq * 3 + k) * n + i];
}

// Farthest point sampling: one block per batch, iterative argmax
__global__ void k_fps(const float* __restrict__ xyz, int n, int npoint,
                      float* __restrict__ dist, int* __restrict__ out)
{
    __shared__ float sv[256];
    __shared__ int   si[256];
    __shared__ int   sfar;
    int bq = blockIdx.x;
    int t  = threadIdx.x;
    const float* P = xyz + (long long)bq * n * 3;
    float* D = dist + (long long)bq * n;
    if (t == 0) sfar = 0;
    for (int i = t; i < n; i += 256) D[i] = 1e10f;
    __syncthreads();
    for (int it = 0; it < npoint; ++it) {
        int far = sfar;
        if (t == 0) out[bq * npoint + it] = far;
        float cx = P[far*3+0], cy = P[far*3+1], cz = P[far*3+2];
        float best = -1.0f; int bi = 0;
        for (int i = t; i < n; i += 256) {
            float dx = P[i*3]-cx, dy = P[i*3+1]-cy, dz = P[i*3+2]-cz;
            float d  = dx*dx + dy*dy + dz*dz;
            float nd = fminf(D[i], d);
            D[i] = nd;
            if (nd > best) { best = nd; bi = i; }
        }
        sv[t] = best; si[t] = bi;
        __syncthreads();
        for (int s2 = 128; s2 > 0; s2 >>= 1) {
            if (t < s2 && sv[t+s2] > sv[t]) { sv[t] = sv[t+s2]; si[t] = si[t+s2]; }
            __syncthreads();
        }
        if (t == 0) sfar = si[0];
        __syncthreads();
    }
}

// dst[q][3] = src[b][idx[q]][3]
__global__ void k_gather_xyz(const float* __restrict__ src, int n,
                             const int* __restrict__ idx, int S,
                             float* __restrict__ dst, int Bn)
{
    int q = blockIdx.x * blockDim.x + threadIdx.x;
    if (q >= Bn * S) return;
    int bq = q / S;
    int i  = idx[q];
    long long s = ((long long)bq * n + i) * 3;
#pragma unroll
    for (int k = 0; k < 3; ++k) dst[q*3+k] = src[s+k];
}

// First nsample(32) indices with d2 <= r2, padded with first hit
__global__ void k_ball(const float* __restrict__ xyz, const float* __restrict__ nxyz,
                       int n, int S, float r2, int* __restrict__ out, int Bn)
{
    int q = blockIdx.x * blockDim.x + threadIdx.x;
    if (q >= Bn * S) return;
    int bq = q / S;
    const float* P = xyz + (long long)bq * n * 3;
    float qx = nxyz[q*3], qy = nxyz[q*3+1], qz = nxyz[q*3+2];
    int* o = out + (long long)q * 32;
    int cnt = 0;
    for (int i = 0; i < n && cnt < 32; ++i) {
        float dx = P[i*3]-qx, dy = P[i*3+1]-qy, dz = P[i*3+2]-qz;
        if (dx*dx + dy*dy + dz*dz <= r2) o[cnt++] = i;
    }
    int first = (cnt > 0) ? o[0] : 0;
    for (int j = cnt; j < 32; ++j) o[j] = first;
}

// Build grouped GEMM panel: out[c][p] = feat[c][gi]-feat[c][fi]; out[Cin+c][p]=feat[c][fi]
__global__ void k_group(const _Float16* __restrict__ feat, int Npts, int S, int Cin,
                        const int* __restrict__ fps, const int* __restrict__ ball,
                        _Float16* __restrict__ out, int Bn)
{
    long long p   = (long long)blockIdx.x * blockDim.x + threadIdx.x;
    long long tot = (long long)Bn * S * 32;
    if (p >= tot) return;
    int bq = (int)(p / ((long long)S * 32));
    int s  = (int)((p / 32) % S);
    int gi = ball[p];
    int fi = fps[bq * S + s];
    long long NP   = (long long)Bn * Npts;
    long long colg = (long long)bq * Npts + gi;
    long long colc = (long long)bq * Npts + fi;
    for (int c = 0; c < Cin; ++c) {
        float fg = (float)feat[(long long)c * NP + colg];
        float fc = (float)feat[(long long)c * NP + colc];
        out[(long long)c * tot + p]         = (_Float16)(fg - fc);
        out[(long long)(Cin + c) * tot + p] = (_Float16)fc;
    }
}

// out[c][g] = max_j in[c][g*32+j]
__global__ void k_maxpool(const _Float16* __restrict__ in, int C, long long G,
                          float* __restrict__ of, _Float16* __restrict__ oh)
{
    long long t = (long long)blockIdx.x * blockDim.x + threadIdx.x;
    if (t >= (long long)C * G) return;
    int c = (int)(t / G); long long g = t % G;
    const _Float16* p = in + (long long)c * G * 32 + g * 32;
    float m = -1e30f;
#pragma unroll
    for (int j = 0; j < 32; ++j) m = fmaxf(m, (float)p[j]);
    if (of) of[t] = m;
    if (oh) oh[t] = (_Float16)m;
}

// elementwise op: 0=add 1=sub; optional f32 + f16 stores
__global__ void k_ew(const float* __restrict__ a, const float* __restrict__ b, int op,
                     float* __restrict__ of, _Float16* __restrict__ oh, long long n)
{
    long long i = (long long)blockIdx.x * blockDim.x + threadIdx.x;
    if (i >= n) return;
    float v = (op == 0) ? (a[i] + b[i]) : (a[i] - b[i]);
    if (of) of[i] = v;
    if (oh) oh[i] = (_Float16)v;
}

// f16 transpose [R][Cc] -> [Cc][R]
__global__ void k_tr16(const _Float16* __restrict__ in, int R, int Cc,
                       _Float16* __restrict__ out)
{
    long long i = (long long)blockIdx.x * blockDim.x + threadIdx.x;
    if (i >= (long long)R * Cc) return;
    int r = (int)(i / Cc); int c = (int)(i % Cc);
    out[(long long)c * R + r] = in[i];
}

// nxyz2[q][3] -> xyzT16[k][q]
__global__ void k_xyzT(const float* __restrict__ nxyz2, _Float16* __restrict__ out, int tot)
{
    int q = blockIdx.x * blockDim.x + threadIdx.x;
    if (q >= tot) return;
#pragma unroll
    for (int k = 0; k < 3; ++k) out[k * tot + q] = (_Float16)nxyz2[q*3+k];
}

// row softmax over 256 columns; one block per row
__global__ void k_softmax_row(float* __restrict__ attn, int Mc)
{
    __shared__ float red[256];
    int row = blockIdx.x; int t = threadIdx.x;
    float* p = attn + (long long)row * Mc;
    float v = p[t];
    red[t] = v; __syncthreads();
    for (int s = 128; s > 0; s >>= 1) { if (t < s) red[t] = fmaxf(red[t], red[t+s]); __syncthreads(); }
    float mx = red[0]; __syncthreads();
    float e = __expf(v - mx);
    red[t] = e; __syncthreads();
    for (int s = 128; s > 0; s >>= 1) { if (t < s) red[t] += red[t+s]; __syncthreads(); }
    p[t] = e / red[0];
}

// column renorm attn/(1e-9+colsum) -> f16
__global__ void k_colnorm(const float* __restrict__ attn, _Float16* __restrict__ out,
                          int Ntok, int total)
{
    int t = blockIdx.x * blockDim.x + threadIdx.x;
    if (t >= total) return;
    int bq = t / Ntok; int m = t % Ntok;
    const float* p = attn + (long long)bq * Ntok * Ntok + m;
    float s = 0.0f;
    for (int nr = 0; nr < Ntok; ++nr) s += p[(long long)nr * Ntok];
    float inv = 1.0f / (1e-9f + s);
    _Float16* o = out + (long long)bq * Ntok * Ntok + m;
    for (int nr = 0; nr < Ntok; ++nr)
        o[(long long)nr * Ntok] = (_Float16)(p[(long long)nr * Ntok] * inv);
}

// max over T tokens: in [C][Bn*T] -> out f16 [C][Bn]
__global__ void k_fusemax(const float* __restrict__ in, int C, int Bn, int T,
                          _Float16* __restrict__ out)
{
    int t = blockIdx.x * blockDim.x + threadIdx.x;
    if (t >= C * Bn) return;
    int c = t / Bn, b = t % Bn;
    const float* p = in + (long long)c * Bn * T + (long long)b * T;
    float m = -1e30f;
    for (int j = 0; j < T; ++j) m = fmaxf(m, p[j]);
    out[c * Bn + b] = (_Float16)m;
}

// fea [128][8] f32 -> d_out: fea(8x128) | sigmoid(cls)(8) | 0.3*off(8x3)
__global__ void k_final(const float* __restrict__ fea,
                        const float* __restrict__ offw, const float* __restrict__ offb,
                        const float* __restrict__ clsw, const float* __restrict__ clsb,
                        float* __restrict__ out)
{
    int b = threadIdx.x;
    if (b >= 8) return;
    for (int c = 0; c < 128; ++c) out[b * 128 + c] = fea[c * 8 + b];
    float s = clsb[0];
    for (int c = 0; c < 128; ++c) s += clsw[c] * fea[c * 8 + b];
    out[1024 + b] = 1.0f / (1.0f + __expf(-s));
    for (int o2 = 0; o2 < 3; ++o2) {
        float so = offb[o2];
        for (int c = 0; c < 128; ++c) so += offw[o2 * 128 + c] * fea[c * 8 + b];
        out[1032 + b * 3 + o2] = 0.3f * so;
    }
}

// ---------------------------------------------------------------------------
// Host orchestration
// ---------------------------------------------------------------------------
extern "C" void kernel_launch(void* const* d_in, const int* in_sizes, int n_in,
                              void* d_out, int out_size, void* d_ws, size_t ws_size,
                              hipStream_t stream)
{
    (void)in_sizes; (void)n_in; (void)out_size; (void)ws_size;

    // ---- parameter indices (setup_inputs dict order) ----
    enum {
        IN_PTS = 0,
        W_C1, BN1G, BN1B, W_C2, BN2G, BN2B,
        W_G0C1, G0B1G, G0B1B, W_G0C2, G0B2G, G0B2B,
        W_G1C1, G1B1G, G1B1B, W_G1C2, G1B2G, G1B2B,
        W_PT, PTBG, PTBB, W_POS, B_POS,
        SA_BASE,                                  // 24: 4 x 7 entries
        W_FUSE = 52, FUSEG, FUSEB,
        W_L1, BN6G, BN6B,
        W_L2, L2B, BN7G, BN7B,
        W_L3, L3B,
        W_OFF, B_OFF, W_CLS, B_CLS
    };
    auto F = [&](int i) -> const float* { return (const float*)d_in[i]; };

    const int Bn = 8, Npts = 16384, S1 = 512, S2 = 256;
    const long long N1 = (long long)Bn * Npts;      // 131072
    const long long G1 = (long long)Bn * S1;        // 4096
    const long long GP0 = G1 * 32;                  // 131072
    const long long G2 = (long long)Bn * S2;        // 2048
    const long long GP1 = G2 * 32;                  // 65536
    const int T = 2048;                             // tokens (B*256) in pt stage

    // ---- workspace bump allocator ----
    char* wsb = (char*)d_ws;
    size_t pos = 0;
    auto alloc = [&](size_t bytes) -> void* {
        pos = (pos + 255) & ~(size_t)255;
        void* p = wsb + pos;
        pos += bytes;
        return p;
    };

    auto cvtN = [&](const float* src, long long n) -> _Float16* {
        _Float16* p = (_Float16*)alloc((size_t)n * 2);
        k_cvt16<<<dim3((unsigned)((n + 255) / 256)), dim3(256), 0, stream>>>(src, p, n);
        return p;
    };
    auto prep = [&](const float* g, const float* bnb, const float* cb, int C,
                    float** ps, float** pb) {
        *ps = (float*)alloc((size_t)C * 4);
        *pb = (float*)alloc((size_t)C * 4);
        k_prep<<<dim3((C + 63) / 64), dim3(64), 0, stream>>>(g, bnb, cb, *ps, *pb, C);
    };
    auto gemm = [&](const _Float16* A, int lda, long long bsA,
                    const _Float16* Bp, int ldb, long long bsB,
                    float* Cf, _Float16* Ch, int ldc, long long bsC,
                    int M, int Np, int K, int batch,
                    const float* sc, const float* bi, int act) {
        dim3 g((unsigned)((Np + 63) / 64), (unsigned)((M + 31) / 32), (unsigned)batch);
        k_gemm_wmma<<<g, dim3(128), 0, stream>>>(A, lda, bsA, Bp, ldb, bsB,
                                                 Cf, Ch, ldc, bsC, M, Np, K, sc, bi, act);
    };

    // ---- f16 weights ----
    _Float16* w_c1   = cvtN(F(W_C1),   64 * 3);
    _Float16* w_c2   = cvtN(F(W_C2),   64 * 64);
    _Float16* w_g0a  = cvtN(F(W_G0C1), 128 * 128);
    _Float16* w_g0b  = cvtN(F(W_G0C2), 128 * 128);
    _Float16* w_g1a  = cvtN(F(W_G1C1), 256 * 256);
    _Float16* w_g1b  = cvtN(F(W_G1C2), 256 * 256);
    _Float16* w_pt   = cvtN(F(W_PT),   256 * 256);
    _Float16* w_pos  = cvtN(F(W_POS),  256 * 3);
    _Float16* w_qk[4], *w_v[4], *w_tr[4];
    for (int i = 0; i < 4; ++i) {
        int base = SA_BASE + i * 7;
        w_qk[i] = cvtN(F(base + 0), 64 * 256);
        w_v[i]  = cvtN(F(base + 1), 256 * 256);
        w_tr[i] = cvtN(F(base + 3), 256 * 256);
    }
    _Float16* w_fuse = cvtN(F(W_FUSE), 1024 * 1280);
    _Float16* w_l1   = cvtN(F(W_L1),   512 * 1024);
    _Float16* w_l2   = cvtN(F(W_L2),   256 * 512);
    _Float16* w_l3   = cvtN(F(W_L3),   128 * 256);

    // ---- folded scale/bias ----
    float *s_bn1, *b_bn1, *s_bn2, *b_bn2, *s_g0a, *b_g0a, *s_g0b, *b_g0b;
    float *s_g1a, *b_g1a, *s_g1b, *b_g1b, *s_pt, *b_pt, *s_pos, *b_pos;
    float *s_fuse, *b_fuse, *s_l1, *b_l1, *s_l2, *b_l2, *s_l3, *b_l3;
    float *s_v[4], *b_v[4], *s_tr[4], *b_tr[4];
    prep(F(BN1G), F(BN1B), nullptr, 64, &s_bn1, &b_bn1);
    prep(F(BN2G), F(BN2B), nullptr, 64, &s_bn2, &b_bn2);
    prep(F(G0B1G), F(G0B1B), nullptr, 128, &s_g0a, &b_g0a);
    prep(F(G0B2G), F(G0B2B), nullptr, 128, &s_g0b, &b_g0b);
    prep(F(G1B1G), F(G1B1B), nullptr, 256, &s_g1a, &b_g1a);
    prep(F(G1B2G), F(G1B2B), nullptr, 256, &s_g1b, &b_g1b);
    prep(F(PTBG), F(PTBB), nullptr, 256, &s_pt, &b_pt);
    prep(nullptr, nullptr, F(B_POS), 256, &s_pos, &b_pos);
    for (int i = 0; i < 4; ++i) {
        int base = SA_BASE + i * 7;
        prep(nullptr, nullptr, F(base + 2), 256, &s_v[i], &b_v[i]);           // v bias
        prep(F(base + 5), F(base + 6), F(base + 4), 256, &s_tr[i], &b_tr[i]); // bn + tr bias
    }
    prep(F(FUSEG), F(FUSEB), nullptr, 1024, &s_fuse, &b_fuse);
    prep(F(BN6G), F(BN6B), nullptr, 512, &s_l1, &b_l1);
    prep(F(BN7G), F(BN7B), F(L2B), 256, &s_l2, &b_l2);
    prep(nullptr, nullptr, F(L3B), 128, &s_l3, &b_l3);

    // ---- persistent buffers ----
    _Float16* pts16 = (_Float16*)alloc((size_t)Bn * 3 * Npts * 2);
    float* xyz1  = (float*)alloc((size_t)Bn * Npts * 3 * 4);
    float* dist  = (float*)alloc((size_t)Bn * Npts * 4);
    int*   fps1  = (int*)alloc((size_t)Bn * S1 * 4);
    float* nxyz  = (float*)alloc((size_t)Bn * S1 * 3 * 4);
    int*   ball1 = (int*)alloc((size_t)Bn * S1 * 32 * 4);
    int*   fps2  = (int*)alloc((size_t)Bn * S2 * 4);
    float* nxyz2 = (float*)alloc((size_t)Bn * S2 * 3 * 4);
    int*   ball2 = (int*)alloc((size_t)Bn * S2 * 32 * 4);
    _Float16* f0_16 = (_Float16*)alloc((size_t)128 * G1 * 2);
    _Float16* cat16 = (_Float16*)alloc((size_t)1280 * T * 2);
    _Float16* f1_16 = cat16 + (size_t)1024 * T;     // rows 1024..1279 of fuse input

    // ---- big phase-1 regions (recycled for phase 2) ----
    size_t big_mark = (pos + 255) & ~(size_t)255;
    pos = big_mark;
    _Float16* feat1 = (_Float16*)alloc((size_t)64 * N1 * 2);
    _Float16* feat2 = (_Float16*)alloc((size_t)64 * N1 * 2);
    _Float16* RC    = (_Float16*)alloc((size_t)128 * GP0 * 2);
    _Float16* RD    = (_Float16*)alloc((size_t)128 * GP0 * 2);

    // ================= Phase 1: stem + grouping + local ops =================
    k_cvt16<<<dim3((unsigned)((Bn * 3LL * Npts + 255) / 256)), dim3(256), 0, stream>>>(
        F(IN_PTS), pts16, (long long)Bn * 3 * Npts);
    k_pts2xyz<<<dim3((unsigned)((Bn * (long long)Npts + 255) / 256)), dim3(256), 0, stream>>>(
        F(IN_PTS), xyz1, Npts);

    // conv1 (batched over b since input is [b][3][n]) -> feat1 [64][B*N] f16
    gemm(w_c1, 3, 0, pts16, Npts, 3LL * Npts,
         nullptr, feat1, (int)N1, Npts, 64, Npts, 3, Bn, s_bn1, b_bn1, 1);
    // conv2 -> feat2
    gemm(w_c2, 64, 0, feat1, (int)N1, 0,
         nullptr, feat2, (int)N1, 0, 64, (int)N1, 64, 1, s_bn2, b_bn2, 1);

    k_fps<<<dim3(Bn), dim3(256), 0, stream>>>(xyz1, Npts, S1, dist, fps1);
    k_gather_xyz<<<dim3((Bn * S1 + 127) / 128), dim3(128), 0, stream>>>(
        xyz1, Npts, fps1, S1, nxyz, Bn);
    k_ball<<<dim3((Bn * S1 + 127) / 128), dim3(128), 0, stream>>>(
        xyz1, nxyz, Npts, S1, 0.15f * 0.15f, ball1, Bn);
    k_group<<<dim3((unsigned)((GP0 + 127) / 128)), dim3(128), 0, stream>>>(
        feat2, Npts, S1, 64, fps1, ball1, RC, Bn);

    // g0: 128x128 convs over 131072 columns
    gemm(w_g0a, 128, 0, RC, (int)GP0, 0, nullptr, RD, (int)GP0, 0,
         128, (int)GP0, 128, 1, s_g0a, b_g0a, 1);
    gemm(w_g0b, 128, 0, RD, (int)GP0, 0, nullptr, RC, (int)GP0, 0,
         128, (int)GP0, 128, 1, s_g0b, b_g0b, 1);
    k_maxpool<<<dim3((unsigned)((128 * G1 + 255) / 256)), dim3(256), 0, stream>>>(
        RC, 128, G1, nullptr, f0_16);

    // second grouping on the 512 sampled points
    k_fps<<<dim3(Bn), dim3(256), 0, stream>>>(nxyz, S1, S2, dist, fps2);
    k_gather_xyz<<<dim3((Bn * S2 + 127) / 128), dim3(128), 0, stream>>>(
        nxyz, S1, fps2, S2, nxyz2, Bn);
    k_ball<<<dim3((Bn * S2 + 127) / 128), dim3(128), 0, stream>>>(
        nxyz, nxyz2, S1, S2, 0.2f * 0.2f, ball2, Bn);
    k_group<<<dim3((unsigned)((GP1 + 127) / 128)), dim3(128), 0, stream>>>(
        f0_16, S1, S2, 128, fps2, ball2, RD, Bn);

    // g1: 256x256 convs over 65536 columns
    gemm(w_g1a, 256, 0, RD, (int)GP1, 0, nullptr, RC, (int)GP1, 0,
         256, (int)GP1, 256, 1, s_g1a, b_g1a, 1);
    gemm(w_g1b, 256, 0, RC, (int)GP1, 0, nullptr, RD, (int)GP1, 0,
         256, (int)GP1, 256, 1, s_g1b, b_g1b, 1);
    k_maxpool<<<dim3((unsigned)((256 * G2 + 255) / 256)), dim3(256), 0, stream>>>(
        RD, 256, G2, nullptr, f1_16);

    // ================= Phase 2: pt_last + fuse + heads (reuse big regions) ==
    pos = big_mark;
    _Float16* xyzT16 = (_Float16*)alloc((size_t)3 * T * 2);
    float* pe    = (float*)alloc((size_t)256 * T * 4);
    float* xbuf  = (float*)alloc((size_t)256 * T * 4);
    float* xinF  = (float*)alloc((size_t)256 * T * 4);
    _Float16* xin16 = (_Float16*)alloc((size_t)256 * T * 2);
    _Float16* QK16  = (_Float16*)alloc((size_t)64 * T * 2);
    _Float16* Qt16  = (_Float16*)alloc((size_t)T * 64 * 2);
    _Float16* v16   = (_Float16*)alloc((size_t)256 * T * 2);
    float* attnF = (float*)alloc((size_t)Bn * 256 * 256 * 4);
    _Float16* attn16 = (_Float16*)alloc((size_t)Bn * 256 * 256 * 2);
    float* xrF   = (float*)alloc((size_t)256 * T * 4);
    _Float16* t16 = (_Float16*)alloc((size_t)256 * T * 2);
    float* xr2F  = (float*)alloc((size_t)256 * T * 4);
    float* fusedF = (float*)alloc((size_t)1024 * T * 4);
    _Float16* xT16 = (_Float16*)alloc((size_t)1024 * 8 * 2);
    _Float16* h1   = (_Float16*)alloc((size_t)512 * 8 * 2);
    _Float16* h2   = (_Float16*)alloc((size_t)256 * 8 * 2);
    float* feaF = (float*)alloc((size_t)128 * 8 * 4);

    k_xyzT<<<dim3((T + 127) / 128), dim3(128), 0, stream>>>(nxyz2, xyzT16, T);
    // positional encoding: 3 -> 256, bias only
    gemm(w_pos, 3, 0, xyzT16, T, 0, pe, nullptr, T, 0, 256, T, 3, 1, s_pos, b_pos, 0);
    // pt_conv1
    gemm(w_pt, 256, 0, f1_16, T, 0, xbuf, nullptr, T, 0, 256, T, 256, 1, s_pt, b_pt, 1);

    const long long ET = 256LL * T;
    for (int i = 0; i < 4; ++i) {
        // xin = x + pe
        k_ew<<<dim3((unsigned)((ET + 255) / 256)), dim3(256), 0, stream>>>(
            xbuf, pe, 0, xinF, xin16, ET);
        // q/k projection (shared weights)
        gemm(w_qk[i], 256, 0, xin16, T, 0, nullptr, QK16, T, 0, 64, T, 256, 1,
             nullptr, nullptr, 0);
        k_tr16<<<dim3((unsigned)((64LL * T + 255) / 256)), dim3(256), 0, stream>>>(
            QK16, 64, T, Qt16);
        // v projection (+bias)
        gemm(w_v[i], 256, 0, xin16, T, 0, nullptr, v16, T, 0, 256, T, 256, 1,
             s_v[i], b_v[i], 0);
        // attn logits: per batch (256x64)x(64x256)
        gemm(Qt16, 64, 256LL * 64, QK16, T, 256, attnF, nullptr, 256, 256LL * 256,
             256, 256, 64, Bn, nullptr, nullptr, 0);
        k_softmax_row<<<dim3(Bn * 256), dim3(256), 0, stream>>>(attnF, 256);
        k_colnorm<<<dim3((Bn * 256 + 127) / 128), dim3(128), 0, stream>>>(
            attnF, attn16, 256, Bn * 256);
        // x_r = v @ attn : per batch (256x256)x(256x256)
        gemm(v16, T, 256, attn16, 256, 256LL * 256, xrF, nullptr, T, 256,
             256, 256, 256, Bn, nullptr, nullptr, 0);
        // t = xin - x_r
        k_ew<<<dim3((unsigned)((ET + 255) / 256)), dim3(256), 0, stream>>>(
            xinF, xrF, 1, nullptr, t16, ET);
        // x_r2 = relu(bn(tr(t)+tr_b))
        gemm(w_tr[i], 256, 0, t16, T, 0, xr2F, nullptr, T, 0, 256, T, 256, 1,
             s_tr[i], b_tr[i], 1);
        // x = xin + x_r2 ; also f16 copy into fuse-input rows
        k_ew<<<dim3((unsigned)((ET + 255) / 256)), dim3(256), 0, stream>>>(
            xinF, xr2F, 0, xbuf, cat16 + (size_t)i * 256 * T, ET);
    }

    // fuse: 1280 -> 1024, leaky 0.2
    gemm(w_fuse, 1280, 0, cat16, T, 0, fusedF, nullptr, T, 0,
         1024, T, 1280, 1, s_fuse, b_fuse, 2);
    k_fusemax<<<dim3((1024 * Bn + 127) / 128), dim3(128), 0, stream>>>(
        fusedF, 1024, Bn, 256, xT16);

    // linear head chain (N = batch = 8 columns)
    gemm(w_l1, 1024, 0, xT16, 8, 0, nullptr, h1, 8, 0, 512, 8, 1024, 1, s_l1, b_l1, 2);
    gemm(w_l2, 512, 0, h1, 8, 0, nullptr, h2, 8, 0, 256, 8, 512, 1, s_l2, b_l2, 2);
    gemm(w_l3, 256, 0, h2, 8, 0, feaF, nullptr, 8, 0, 128, 8, 256, 1, s_l3, b_l3, 0);

    k_final<<<dim3(1), dim3(8), 0, stream>>>(feaF, F(W_OFF), F(B_OFF),
                                             F(W_CLS), F(B_CLS), (float*)d_out);
}
